// TransformerBlock_Attention_Chosen_QMLP_41008347742973
// MI455X (gfx1250) — matrix operand order repaired and checked by reading, not compiled, not run on hardware
//
#include <hip/hip_runtime.h>

typedef __bf16 bf16;
typedef bf16  v16bf __attribute__((ext_vector_type(16)));
typedef bf16  v8bf  __attribute__((ext_vector_type(8)));
typedef float v8f   __attribute__((ext_vector_type(8)));
typedef int   v4i   __attribute__((ext_vector_type(4)));

#define B_  2
#define S_  1024
#define E_  768
#define H_  12
#define D_  64
#define M_  768
#define Q_  4
#define R_  (B_*S_)      // 2048 token rows
#define QM_ (Q_*M_)      // 3072

// GEMM macro-tile: block = 8 waves; each wave 32x64 (2x4 WMMA tiles); block 256x64
#define BM 256
#define BN 64
#define BK 32

#ifndef __has_builtin
#define __has_builtin(x) 0
#endif
#if __has_builtin(__builtin_amdgcn_global_load_async_to_lds_b128)
#define HAVE_ASYNC 1
typedef __attribute__((address_space(1))) v4i gas_v4i;  // global
typedef __attribute__((address_space(3))) v4i las_v4i;  // LDS
#else
#define HAVE_ASYNC 0
#endif

// ---------- bf16 helpers (round-to-nearest-even) ----------
static __device__ __forceinline__ float bf2f(bf16 h) {
  unsigned short s = __builtin_bit_cast(unsigned short, h);
  unsigned int u = ((unsigned int)s) << 16;
  return __builtin_bit_cast(float, u);
}
static __device__ __forceinline__ bf16 f2bf(float f) {
  unsigned int u = __builtin_bit_cast(unsigned int, f);
  u += 0x7fffu + ((u >> 16) & 1u);
  unsigned short s = (unsigned short)(u >> 16);
  return __builtin_bit_cast(bf16, s);
}

// ---------- WMMA fragment loaders (16-bit, 16x16x32) ----------
// A (16x32, MxK): lane l<16 -> row l, K {0..7}+{16..23}; lane l>=16 -> row l-16, K {8..15}+{24..31}
static __device__ __forceinline__ v16bf load_frag_a(const bf16* base, int ld, int lane) {
  int row = lane & 15;
  int k0  = (lane & 16) ? 8 : 0;
  const bf16* p = base + (size_t)row * ld + k0;
  v8bf lo = *(const v8bf*)(p);
  v8bf hi = *(const v8bf*)(p + 16);
  v16bf r;
#pragma unroll
  for (int i = 0; i < 8; ++i) { r[i] = lo[i]; r[i + 8] = hi[i]; }
  return r;
}
// B (32x16, KxN) built from row-major W[N,K]: lane n<16 -> col n, K {0..7}+{16..23}; lane n>=16 -> col n-16, K {8..15}+{24..31}
static __device__ __forceinline__ v16bf load_frag_b(const bf16* base, int ld, int lane) {
  int col = lane & 15;
  int k0  = (lane & 16) ? 8 : 0;
  const bf16* p = base + (size_t)col * ld + k0;
  v8bf lo = *(const v8bf*)(p);
  v8bf hi = *(const v8bf*)(p + 16);
  v16bf r;
#pragma unroll
  for (int i = 0; i < 8; ++i) { r[i] = lo[i]; r[i + 8] = hi[i]; }
  return r;
}

// ---------- async weight-slab staging (W[n0..n0+63][k..k+31] -> LDS, col-major rows of 32) ----------
static __device__ __forceinline__ void stage_slab(const bf16* __restrict__ W, int ldb,
                                                  int n0, int k, bf16* slab) {
  int idx  = threadIdx.x;            // 0..255: 64 cols x 4 x 16B
  int col  = idx >> 2;
  int part = (idx & 3) * 8;
  const bf16* g = W + (size_t)(n0 + col) * ldb + k + part;
  bf16* l = slab + col * BK + part;
#if HAVE_ASYNC
  __builtin_amdgcn_global_load_async_to_lds_b128((gas_v4i*)g, (las_v4i*)l, 0, 0);
#else
  *(v8bf*)l = *(const v8bf*)g;
#endif
}
static __device__ __forceinline__ void wait_async_done() {
#if HAVE_ASYNC
#if __has_builtin(__builtin_amdgcn_s_wait_asynccnt)
  __builtin_amdgcn_s_wait_asynccnt(0);
#else
  asm volatile("s_wait_asynccnt 0" ::: "memory");
#endif
#endif
}

// ---------- block GEMM core: each wave accumulates 2x4 tiles of C = A * W^T ----------
static __device__ __forceinline__ void gemm_block_acc(const bf16* __restrict__ A, int lda,
                                                      const bf16* __restrict__ W, int ldb,
                                                      int K, int m0, int n0,
                                                      bf16* slab, v8f acc[2][4]) {
  const int lane  = threadIdx.x & 31;
  const int wv    = threadIdx.x >> 5;
  const int mwave = m0 + wv * 32;
  v8f zero = {};
#pragma unroll
  for (int mi = 0; mi < 2; ++mi)
#pragma unroll
    for (int ni = 0; ni < 4; ++ni) acc[mi][ni] = zero;

  for (int k = 0; k < K; k += BK) {
    stage_slab(W, ldb, n0, k, slab);           // block-shared weight slab (async -> LDS)
    wait_async_done();
    __syncthreads();
    v16bf af[2];
    af[0] = load_frag_a(A + (size_t)mwave * lda + k, lda, lane);
    af[1] = load_frag_a(A + (size_t)(mwave + 16) * lda + k, lda, lane);
    __builtin_prefetch(A + (size_t)(mwave + (lane & 15)) * lda + k + BK, 0, 3);
    v16bf bfr[4];
#pragma unroll
    for (int ni = 0; ni < 4; ++ni) bfr[ni] = load_frag_b(slab + (size_t)(ni * 16) * BK, BK, lane);
#pragma unroll
    for (int mi = 0; mi < 2; ++mi)
#pragma unroll
      for (int ni = 0; ni < 4; ++ni)
        acc[mi][ni] = __builtin_amdgcn_wmma_f32_16x16x32_bf16(
            false, af[mi], false, bfr[ni], (short)0, acc[mi][ni], false, false);
    __syncthreads();
  }
}

// ---------- elementwise + LN kernels ----------
__global__ void __launch_bounds__(256) k_f32_to_bf16(const float* __restrict__ in,
                                                     bf16* __restrict__ out, int n) {
  int i = blockIdx.x * 256 + threadIdx.x;
  if (i < n) out[i] = f2bf(in[i]);
}

__global__ void __launch_bounds__(256) k_layernorm(const float* __restrict__ x,
                                                   const float* __restrict__ w,
                                                   const float* __restrict__ bb,
                                                   bf16* __restrict__ out) {
  __shared__ float red[8];
  int row = blockIdx.x, tid = threadIdx.x;
  const float* xr = x + (size_t)row * E_;
  float v[3], s = 0.f;
#pragma unroll
  for (int i = 0; i < 3; ++i) { v[i] = xr[tid + 256 * i]; s += v[i]; }
#pragma unroll
  for (int m = 1; m < 32; m <<= 1) s += __shfl_xor(s, m, 32);
  if ((tid & 31) == 0) red[tid >> 5] = s;
  __syncthreads();
  float tot = 0.f;
#pragma unroll
  for (int i = 0; i < 8; ++i) tot += red[i];
  float mu = tot * (1.f / E_);
  __syncthreads();
  float vs = 0.f;
#pragma unroll
  for (int i = 0; i < 3; ++i) { float d = v[i] - mu; vs += d * d; }
#pragma unroll
  for (int m = 1; m < 32; m <<= 1) vs += __shfl_xor(vs, m, 32);
  if ((tid & 31) == 0) red[tid >> 5] = vs;
  __syncthreads();
  float tv = 0.f;
#pragma unroll
  for (int i = 0; i < 8; ++i) tv += red[i];
  float rs = rsqrtf(tv * (1.f / E_) + 1e-5f);
#pragma unroll
  for (int i = 0; i < 3; ++i) {
    int c = tid + 256 * i;
    out[(size_t)row * E_ + c] = f2bf((v[i] - mu) * rs * w[c] + bb[c]);
  }
}

// ---------- GEMM epilogue variants (grid: x = N/BN, y = Mr/BM) ----------
#define EPILOGUE_HEAD                                        \
  const int lane = threadIdx.x & 31, wv = threadIdx.x >> 5;  \
  const int nn = lane & 15, hi8 = (lane >> 4) << 3;          \
  (void)nn; (void)hi8;

__global__ void __launch_bounds__(256)
k_gemm_bias_bf16(const bf16* __restrict__ A, int lda, const bf16* __restrict__ W,
                 const float* __restrict__ bias, bf16* __restrict__ out, int N, int K) {
  __shared__ bf16 slab[BN * BK];
  v8f acc[2][4];
  const int m0 = blockIdx.y * BM, n0 = blockIdx.x * BN;
  gemm_block_acc(A, lda, W, K, K, m0, n0, slab, acc);
  EPILOGUE_HEAD
#pragma unroll
  for (int mi = 0; mi < 2; ++mi)
#pragma unroll
    for (int ni = 0; ni < 4; ++ni)
#pragma unroll
      for (int r = 0; r < 8; ++r) {
        int m = m0 + wv * 32 + mi * 16 + hi8 + r;
        int n = n0 + ni * 16 + nn;
        out[(size_t)m * N + n] = f2bf(acc[mi][ni][r] + bias[n]);
      }
}

// V projection, written transposed per head: vT[(b*H+h)*D + d][S]
__global__ void __launch_bounds__(256)
k_gemm_bias_vT(const bf16* __restrict__ A, int lda, const bf16* __restrict__ W,
               const float* __restrict__ bias, bf16* __restrict__ vT, int N, int K) {
  __shared__ bf16 slab[BN * BK];
  v8f acc[2][4];
  const int m0 = blockIdx.y * BM, n0 = blockIdx.x * BN;
  gemm_block_acc(A, lda, W, K, K, m0, n0, slab, acc);
  EPILOGUE_HEAD
#pragma unroll
  for (int mi = 0; mi < 2; ++mi)
#pragma unroll
    for (int ni = 0; ni < 4; ++ni)
#pragma unroll
      for (int r = 0; r < 8; ++r) {
        int m = m0 + wv * 32 + mi * 16 + hi8 + r;
        int n = n0 + ni * 16 + nn;
        int b = m / S_, s = m % S_, h = n / D_, d = n % D_;
        vT[((size_t)(b * H_ + h) * D_ + d) * S_ + s] = f2bf(acc[mi][ni][r] + bias[n]);
      }
}

// bias + fp32 residual -> fp32 out (Wo projection and fc2)
__global__ void __launch_bounds__(256)
k_gemm_bias_res_f32(const bf16* __restrict__ A, int lda, const bf16* __restrict__ W,
                    const float* __restrict__ bias, const float* __restrict__ res,
                    float* __restrict__ out, int N, int K) {
  __shared__ bf16 slab[BN * BK];
  v8f acc[2][4];
  const int m0 = blockIdx.y * BM, n0 = blockIdx.x * BN;
  gemm_block_acc(A, lda, W, K, K, m0, n0, slab, acc);
  EPILOGUE_HEAD
#pragma unroll
  for (int mi = 0; mi < 2; ++mi)
#pragma unroll
    for (int ni = 0; ni < 4; ++ni)
#pragma unroll
      for (int r = 0; r < 8; ++r) {
        int m = m0 + wv * 32 + mi * 16 + hi8 + r;
        int n = n0 + ni * 16 + nn;
        out[(size_t)m * N + n] = acc[mi][ni][r] + bias[n] + res[(size_t)m * N + n];
      }
}

// fc1: bias, scatter column n into the <=4 overlapping slice buffers h1s[q][R][M]
__global__ void __launch_bounds__(256)
k_gemm_fc1_slices(const bf16* __restrict__ A, int lda, const bf16* __restrict__ W,
                  const float* __restrict__ bias, bf16* __restrict__ h1s, int N, int K) {
  __shared__ bf16 slab[BN * BK];
  v8f acc[2][4];
  const int m0 = blockIdx.y * BM, n0 = blockIdx.x * BN;
  gemm_block_acc(A, lda, W, K, K, m0, n0, slab, acc);
  EPILOGUE_HEAD
  const size_t SL = (size_t)R_ * M_;
#pragma unroll
  for (int mi = 0; mi < 2; ++mi)
#pragma unroll
    for (int ni = 0; ni < 4; ++ni)
#pragma unroll
      for (int r = 0; r < 8; ++r) {
        int m = m0 + wv * 32 + mi * 16 + hi8 + r;
        int n = n0 + ni * 16 + nn;
        bf16 v = f2bf(acc[mi][ni][r] + bias[n]);
#pragma unroll
        for (int qq = 0; qq < Q_; ++qq) {
          int j = n - qq;
          if (j >= 0 && j < M_) h1s[qq * SL + (size_t)m * M_ + j] = v;
        }
      }
}

// vqc slice q: bias + exact GELU, interleaved store z[m][n*Q+q]
__global__ void __launch_bounds__(256)
k_gemm_vqc(const bf16* __restrict__ A, int lda, const bf16* __restrict__ W,
           const float* __restrict__ bias, bf16* __restrict__ z, int qslice, int N, int K) {
  __shared__ bf16 slab[BN * BK];
  v8f acc[2][4];
  const int m0 = blockIdx.y * BM, n0 = blockIdx.x * BN;
  gemm_block_acc(A, lda, W, K, K, m0, n0, slab, acc);
  EPILOGUE_HEAD
#pragma unroll
  for (int mi = 0; mi < 2; ++mi)
#pragma unroll
    for (int ni = 0; ni < 4; ++ni)
#pragma unroll
      for (int r = 0; r < 8; ++r) {
        int m = m0 + wv * 32 + mi * 16 + hi8 + r;
        int n = n0 + ni * 16 + nn;
        float x = acc[mi][ni][r] + bias[n];
        float g = 0.5f * x * (1.f + erff(x * 0.70710678118654752f));
        z[(size_t)m * QM_ + n * Q_ + qslice] = f2bf(g);
      }
}

// ---------- attention: flash-style, softmax + RBF blend ----------
__global__ void __launch_bounds__(256)
k_attention(const bf16* __restrict__ qb, const bf16* __restrict__ kb,
            const bf16* __restrict__ vT, const float* __restrict__ pond,
            bf16* __restrict__ vals) {
  __shared__ bf16 lds[8][2][16 * 32];   // per-wave P tiles (softmax / rbf)
  const int lane = threadIdx.x & 31;
  const int wv   = threadIdx.x >> 5;
  const int bh   = blockIdx.x;
  const int b = bh / H_, h = bh % H_;
  const int qt = blockIdx.y * 8 + wv;   // 64 query tiles of 16
  const int m0 = b * S_ + qt * 16;
  const int hi8 = (lane >> 4) << 3;

  const bf16* qbase = qb + (size_t)m0 * E_ + h * D_;
  v16bf a0 = load_frag_a(qbase, E_, lane);
  v16bf a1 = load_frag_a(qbase + 32, E_, lane);

  float qn = 0.f;
#pragma unroll
  for (int i = 0; i < 16; ++i) {
    float u = bf2f(a0[i]), w = bf2f(a1[i]);
    qn += u * u + w * w;
  }
  qn += __shfl_xor(qn, 16, 32);
  float qn_r[8];
#pragma unroll
  for (int r = 0; r < 8; ++r) qn_r[r] = __shfl(qn, r + hi8, 32);

  float mrun[8], Zs[8], Zr[8];
#pragma unroll
  for (int r = 0; r < 8; ++r) { mrun[r] = -1e30f; Zs[r] = 0.f; Zr[r] = 0.f; }
  v8f acc_s[4] = {}, acc_r[4] = {};

  for (int kt = 0; kt < S_ / 32; ++kt) {
    const int k0 = kt * 32;
    v8f c[2] = {};
    float kn[2];
#pragma unroll
    for (int nt = 0; nt < 2; ++nt) {
      const bf16* kbase = kb + (size_t)(b * S_ + k0 + nt * 16) * E_ + h * D_;
      v16bf b0 = load_frag_b(kbase, E_, lane);
      v16bf b1 = load_frag_b(kbase + 32, E_, lane);
      c[nt] = __builtin_amdgcn_wmma_f32_16x16x32_bf16(false, a0, false, b0, (short)0, c[nt], false, false);
      c[nt] = __builtin_amdgcn_wmma_f32_16x16x32_bf16(false, a1, false, b1, (short)0, c[nt], false, false);
      float s2 = 0.f;
#pragma unroll
      for (int i = 0; i < 16; ++i) {
        float u = bf2f(b0[i]), w = bf2f(b1[i]);
        s2 += u * u + w * w;
      }
      s2 += __shfl_xor(s2, 16, 32);
      kn[nt] = s2;
    }
#pragma unroll
    for (int r = 0; r < 8; ++r) {
      float lv[2], rv[2];
      float mx = -1e30f;
#pragma unroll
      for (int nt = 0; nt < 2; ++nt) {
        float dot = c[nt][r];
        lv[nt] = dot * 0.125f;  // / sqrt(64)
        float d2 = fmaxf(qn_r[r] + kn[nt] - 2.f * dot, 0.f);
        float s2 = fminf(fmaxf(qn_r[r], 1e-8f), 1e4f);
        rv[nt] = expf(-d2 / s2);
        mx = fmaxf(mx, lv[nt]);
      }
#pragma unroll
      for (int msk = 1; msk < 16; msk <<= 1) mx = fmaxf(mx, __shfl_xor(mx, msk, 32));
      float m_new = fmaxf(mrun[r], mx);
      float scale = expf(mrun[r] - m_new);
      mrun[r] = m_new;
      float ps = 0.f, pr = 0.f;
#pragma unroll
      for (int nt = 0; nt < 2; ++nt) {
        lv[nt] = expf(lv[nt] - m_new);
        ps += lv[nt]; pr += rv[nt];
      }
#pragma unroll
      for (int msk = 1; msk < 16; msk <<= 1) {
        ps += __shfl_xor(ps, msk, 32);
        pr += __shfl_xor(pr, msk, 32);
      }
      Zs[r] = Zs[r] * scale + ps;
      Zr[r] += pr;
#pragma unroll
      for (int t = 0; t < 4; ++t) acc_s[t][r] *= scale;
      int row = r + hi8, nn = lane & 15;
#pragma unroll
      for (int nt = 0; nt < 2; ++nt) {
        lds[wv][0][row * 32 + nt * 16 + nn] = f2bf(lv[nt]);
        lds[wv][1][row * 32 + nt * 16 + nn] = f2bf(rv[nt]);
      }
    }
    asm volatile("s_wait_dscnt 0" ::: "memory");  // C-layout -> A-layout relayout hazard
    v16bf pa_s = load_frag_a(&lds[wv][0][0], 32, lane);
    v16bf pa_r = load_frag_a(&lds[wv][1][0], 32, lane);
#pragma unroll
    for (int t = 0; t < 4; ++t) {
      const bf16* vb = vT + ((size_t)bh * D_ + t * 16) * S_ + k0;  // contiguous keys per lane
      v16bf vf = load_frag_b(vb, S_, lane);
      acc_s[t] = __builtin_amdgcn_wmma_f32_16x16x32_bf16(false, pa_s, false, vf, (short)0, acc_s[t], false, false);
      acc_r[t] = __builtin_amdgcn_wmma_f32_16x16x32_bf16(false, pa_r, false, vf, (short)0, acc_r[t], false, false);
    }
  }

  float sg = 1.f / (1.f + expf(-pond[0]));
  float p0 = 1.f - sg, p1 = sg, inv = 1.f / (p0 + p1 + 1e-7f);
  int nn = lane & 15;
#pragma unroll
  for (int t = 0; t < 4; ++t) {
#pragma unroll
    for (int r = 0; r < 8; ++r) {
      float val = inv * (p0 * acc_s[t][r] / fmaxf(Zs[r], 1e-30f) +
                         p1 * acc_r[t][r] / fmaxf(Zr[r], 1e-8f));
      vals[(size_t)(m0 + r + hi8) * E_ + h * D_ + t * 16 + nn] = f2bf(val);
    }
  }
}

// ---------- host ----------
extern "C" void kernel_launch(void* const* d_in, const int* in_sizes, int n_in,
                              void* d_out, int out_size, void* d_ws, size_t ws_size,
                              hipStream_t stream) {
  (void)in_sizes; (void)n_in; (void)out_size; (void)ws_size;
  const float* x     = (const float*)d_in[0];
  const float* ln1_w = (const float*)d_in[1];
  const float* ln1_b = (const float*)d_in[2];
  const float* Wq    = (const float*)d_in[3];
  const float* bq    = (const float*)d_in[4];
  const float* Wk    = (const float*)d_in[5];
  const float* bk    = (const float*)d_in[6];
  const float* Wv    = (const float*)d_in[7];
  const float* bv    = (const float*)d_in[8];
  const float* Wo    = (const float*)d_in[9];
  const float* bo    = (const float*)d_in[10];
  const float* pond  = (const float*)d_in[11];
  const float* ln2_w = (const float*)d_in[12];
  const float* ln2_b = (const float*)d_in[13];
  const float* fc1_W = (const float*)d_in[14];
  const float* fc1_b = (const float*)d_in[15];
  const float* vqc_W = (const float*)d_in[16];
  const float* vqc_b = (const float*)d_in[17];
  const float* fc2_W = (const float*)d_in[18];
  const float* fc2_b = (const float*)d_in[19];
  float* out = (float*)d_out;

  char* p = (char*)d_ws;
  auto alloc = [&](size_t bytes) -> void* {
    void* r = (void*)p;
    p += (bytes + 255) & ~(size_t)255;
    return r;
  };
  bf16* wq  = (bf16*)alloc((size_t)E_ * E_ * 2);
  bf16* wk  = (bf16*)alloc((size_t)E_ * E_ * 2);
  bf16* wv  = (bf16*)alloc((size_t)E_ * E_ * 2);
  bf16* wo  = (bf16*)alloc((size_t)E_ * E_ * 2);
  bf16* wf1 = (bf16*)alloc((size_t)QM_ * E_ * 2);
  bf16* wvq = (bf16*)alloc((size_t)M_ * M_ * 2);
  bf16* wf2 = (bf16*)alloc((size_t)E_ * QM_ * 2);
  bf16* hbf = (bf16*)alloc((size_t)R_ * E_ * 2);    // LN out (reused for LN2)
  bf16* qb  = (bf16*)alloc((size_t)R_ * E_ * 2);
  bf16* kb  = (bf16*)alloc((size_t)R_ * E_ * 2);
  bf16* vT  = (bf16*)alloc((size_t)R_ * E_ * 2);    // [B*H, D, S]
  bf16* vls = (bf16*)alloc((size_t)R_ * E_ * 2);
  float* x1 = (float*)alloc((size_t)R_ * E_ * 4);
  bf16* h1s = (bf16*)alloc((size_t)Q_ * R_ * M_ * 2);
  bf16* z   = (bf16*)alloc((size_t)R_ * QM_ * 2);

  auto cvt = [&](const float* src, bf16* dst, int n) {
    k_f32_to_bf16<<<(n + 255) / 256, 256, 0, stream>>>(src, dst, n);
  };
  cvt(Wq, wq, E_ * E_);      cvt(Wk, wk, E_ * E_);
  cvt(Wv, wv, E_ * E_);      cvt(Wo, wo, E_ * E_);
  cvt(fc1_W, wf1, QM_ * E_); cvt(vqc_W, wvq, M_ * M_);
  cvt(fc2_W, wf2, E_ * QM_);

  const dim3 gE(E_ / BN, R_ / BM);    // N=768 GEMMs
  const dim3 gF1(QM_ / BN, R_ / BM);  // N=3072 GEMM

  // attention sublayer
  k_layernorm<<<R_, 256, 0, stream>>>(x, ln1_w, ln1_b, hbf);
  k_gemm_bias_bf16<<<gE, 256, 0, stream>>>(hbf, E_, wq, bq, qb, E_, E_);
  k_gemm_bias_bf16<<<gE, 256, 0, stream>>>(hbf, E_, wk, bk, kb, E_, E_);
  k_gemm_bias_vT  <<<gE, 256, 0, stream>>>(hbf, E_, wv, bv, vT, E_, E_);
  k_attention<<<dim3(B_ * H_, 8), 256, 0, stream>>>(qb, kb, vT, pond, vls);
  k_gemm_bias_res_f32<<<gE, 256, 0, stream>>>(vls, E_, wo, bo, x, x1, E_, E_);

  // sliced MLP sublayer
  k_layernorm<<<R_, 256, 0, stream>>>(x1, ln2_w, ln2_b, hbf);
  k_gemm_fc1_slices<<<gF1, 256, 0, stream>>>(hbf, E_, wf1, fc1_b, h1s, QM_, E_);
  const size_t SL = (size_t)R_ * M_;
  for (int qq = 0; qq < Q_; ++qq)
    k_gemm_vqc<<<gE, 256, 0, stream>>>(h1s + qq * SL, M_, wvq, vqc_b, z, qq, M_, M_);
  k_gemm_bias_res_f32<<<gE, 256, 0, stream>>>(z, QM_, wf2, fc2_b, x1, out, E_, QM_);
}
